// MultiHeadAttention_72052371357999
// MI455X (gfx1250) — matrix operand order repaired and checked by reading, not compiled, not run on hardware
//
#include <hip/hip_runtime.h>

// ---------------------------------------------------------------------------
// MultiHeadAttention for MI455X (gfx1250): bf16 WMMA + TDM async tile loads.
// B=16, S=512, H=768, NH=12, DK=64.
// ---------------------------------------------------------------------------

typedef __attribute__((ext_vector_type(16))) __bf16 v16bf;
typedef __attribute__((ext_vector_type(8)))  float  v8f;
typedef __attribute__((ext_vector_type(4)))  unsigned int v4u;
typedef __attribute__((ext_vector_type(8)))  int    v8i;
typedef __attribute__((ext_vector_type(4)))  int    v4i;

#if __has_builtin(__builtin_amdgcn_tensor_load_to_lds) && \
    __has_builtin(__builtin_amdgcn_s_wait_tensorcnt)
#define USE_TDM 1
#else
#define USE_TDM 0
#endif

union BF16x16 {
    v16bf v;
    unsigned int u[8];
    unsigned short h[16];
};

__device__ __forceinline__ unsigned short f2bf(float f) {
    union { float f; unsigned int u; } x;
    x.f = f;
    unsigned int u = x.u;
    unsigned int r = (u + 0x7FFFu + ((u >> 16) & 1u)) >> 16;  // RNE bf16
    return (unsigned short)r;
}

// 16-bit A-matrix 16x32 K offsets (ISA 7.12.2): lanes 0-15 hold K {0..7,16..23},
// lanes 16-31 hold K {8..15,24..31}; consecutive K pairs packed per VGPR.
__device__ __forceinline__ int a_koff(int j, int hi) {
    return (j < 4) ? (hi * 8 + j * 2) : (16 + hi * 8 + (j - 4) * 2);
}
// 16-bit B-matrix 32x16 K offsets (same as A): lanes 0-15 hold K {0..7,16..23},
// lanes 16-31 hold K {8..15,24..31}.
__device__ __forceinline__ int b_koff(int j, int hi) {
    return (j < 4) ? (hi * 8 + j * 2) : (16 + hi * 8 + (j - 4) * 2);
}

#if USE_TDM
// ---------------------------------------------------------------------------
// TDM: async DMA of a 2D bf16 tile (tile_rows x tile_cols, row stride in
// elements) from global memory into LDS. Issued once per wave; tracked by
// TENSORcnt. D# layout per CDNA5 ISA ch.8 (group0/group1, 2D tensor).
// ---------------------------------------------------------------------------
__device__ __forceinline__ void tdm_load_2d_bf16(const void* gptr, void* ldsptr,
                                                 int tile_rows, int tile_cols,
                                                 int row_stride_elems) {
    unsigned long long ga = (unsigned long long)gptr;
    unsigned lds = (unsigned)(size_t)ldsptr;   // low 32 bits of flat = LDS addr
    v4u g0;
    g0[0] = 1u;                                       // count=1, user mode
    g0[1] = lds;                                      // lds_addr [63:32]
    g0[2] = (unsigned)(ga & 0xFFFFFFFFu);             // global_addr low
    g0[3] = (unsigned)((ga >> 32) & 0x1FFFFFFu)       // global_addr[56:32]
            | (2u << 30);                             // type=2 ("image")
    v8i g1;
    g1[0] = (int)(1u << 16);                          // data_size=1 -> 2 bytes
    g1[1] = (int)((unsigned)tile_cols << 16);         // tensor_dim0 lo16 @48
    g1[2] = (int)((unsigned)tile_rows << 16);         // tensor_dim1 lo16 @80
    g1[3] = (int)((unsigned)tile_cols << 16);         // tile_dim0 @112
    g1[4] = (int)(unsigned)tile_rows;                 // tile_dim1 @128 (dim2=0)
    g1[5] = row_stride_elems;                         // tensor_dim0_stride lo32
    g1[6] = 0;                                        // stride hi + dim1_stride
    g1[7] = 0;
    v4i z4 = {0, 0, 0, 0};
#if defined(__clang_major__) && (__clang_major__ >= 23)
    v8i z8 = {0, 0, 0, 0, 0, 0, 0, 0};
    __builtin_amdgcn_tensor_load_to_lds(g0, g1, z4, z4, z8, 0);
#else
    __builtin_amdgcn_tensor_load_to_lds(g0, g1, z4, z4, 0);
#endif
}
#endif  // USE_TDM

// ---------------------------------------------------------------------------
// fp32 -> bf16 elementwise conversion
// ---------------------------------------------------------------------------
__global__ void f32_to_bf16_kernel(const float* __restrict__ in,
                                   unsigned short* __restrict__ out, int n) {
    int i = blockIdx.x * blockDim.x + threadIdx.x;
    if (i < n) out[i] = f2bf(in[i]);
}

// ---------------------------------------------------------------------------
// GEMM: C[M,N] = A[M,K] (bf16) * W[N,K]^T (bf16) + bias[N]
// 256 threads = 8 waves; block tile 128(M) x 64(N); K-step 32 (one WMMA K).
// Double-buffered LDS tiles, prefetched by TDM (wave 0) one K-step ahead.
// out_f32 != null -> fp32 output, else bf16 output.
// ---------------------------------------------------------------------------
__global__ __launch_bounds__(256)
void gemm_bf16_kernel(const unsigned short* __restrict__ A,
                      const unsigned short* __restrict__ W,
                      const float* __restrict__ bias,
                      unsigned short* __restrict__ out_bf,
                      float* __restrict__ out_f32,
                      int M, int N, int K) {
    __shared__ unsigned short As[2][128 * 32];
    __shared__ unsigned short Bs[2][64 * 32];

    const int tid  = threadIdx.x;
    const int lane = tid & 31;
    const int wave = tid >> 5;
    const int hl   = lane & 15;
    const int hi   = (lane >> 4) & 1;
    const int wm   = wave >> 1;   // 0..3
    const int wn   = wave & 1;    // 0..1

    const int nTilesN = N / 64;
    const int m0 = (blockIdx.x / nTilesN) * 128;
    const int n0 = (blockIdx.x % nTilesN) * 64;

    v8f acc[2][2];
#pragma unroll
    for (int i = 0; i < 2; ++i)
#pragma unroll
        for (int j = 0; j < 2; ++j)
#pragma unroll
            for (int e = 0; e < 8; ++e) acc[i][j][e] = 0.0f;

    const int nIter = K / 32;

#if USE_TDM
    if (wave == 0) {
        tdm_load_2d_bf16(A + (size_t)m0 * K, &As[0][0], 128, 32, K);
        tdm_load_2d_bf16(W + (size_t)n0 * K, &Bs[0][0], 64, 32, K);
    }
#else
    {   // cooperative stage of buffer 0
        int r = tid >> 1, hh = tid & 1;
        const uint4* s = (const uint4*)(A + (size_t)(m0 + r) * K + hh * 16);
        uint4* d = (uint4*)(&As[0][0] + r * 32 + hh * 16);
        d[0] = s[0];
        d[1] = s[1];
        int rb = tid >> 2, q = tid & 3;
        *(uint4*)(&Bs[0][0] + rb * 32 + q * 8) =
            *(const uint4*)(W + (size_t)(n0 + rb) * K + q * 8);
    }
#endif

    for (int it = 0; it < nIter; ++it) {
        const int buf = it & 1;
        const int k0 = it * 32;
#if USE_TDM
        if (wave == 0) __builtin_amdgcn_s_wait_tensorcnt(0);
        __syncthreads();
        if (it + 1 < nIter && wave == 0) {
            tdm_load_2d_bf16(A + (size_t)m0 * K + k0 + 32, &As[buf ^ 1][0],
                             128, 32, K);
            tdm_load_2d_bf16(W + (size_t)n0 * K + k0 + 32, &Bs[buf ^ 1][0],
                             64, 32, K);
        }
#else
        __syncthreads();
        if (it + 1 < nIter) {
            int r = tid >> 1, hh = tid & 1;
            const uint4* s =
                (const uint4*)(A + (size_t)(m0 + r) * K + k0 + 32 + hh * 16);
            uint4* d = (uint4*)(&As[buf ^ 1][0] + r * 32 + hh * 16);
            d[0] = s[0];
            d[1] = s[1];
            int rb = tid >> 2, q = tid & 3;
            *(uint4*)(&Bs[buf ^ 1][0] + rb * 32 + q * 8) =
                *(const uint4*)(W + (size_t)(n0 + rb) * K + k0 + 32 + q * 8);
        }
#endif

        BF16x16 afrag[2], bfrag[2];
#pragma unroll
        for (int tm = 0; tm < 2; ++tm)
#pragma unroll
            for (int j = 0; j < 8; ++j)
                afrag[tm].u[j] = *(const unsigned int*)
                    (&As[buf][0] + (wm * 32 + tm * 16 + hl) * 32 + a_koff(j, hi));
#pragma unroll
        for (int tn = 0; tn < 2; ++tn)
#pragma unroll
            for (int j = 0; j < 8; ++j)
                bfrag[tn].u[j] = *(const unsigned int*)
                    (&Bs[buf][0] + (wn * 32 + tn * 16 + hl) * 32 + b_koff(j, hi));

#pragma unroll
        for (int tm = 0; tm < 2; ++tm)
#pragma unroll
            for (int tn = 0; tn < 2; ++tn)
                acc[tm][tn] = __builtin_amdgcn_wmma_f32_16x16x32_bf16(
                    false, afrag[tm].v, false, bfrag[tn].v,
                    (short)0, acc[tm][tn], false, false);
    }

#pragma unroll
    for (int tm = 0; tm < 2; ++tm)
#pragma unroll
        for (int tn = 0; tn < 2; ++tn) {
            int n = n0 + wn * 32 + tn * 16 + hl;
            float bv = bias ? bias[n] : 0.0f;
#pragma unroll
            for (int r = 0; r < 8; ++r) {
                int m = m0 + wm * 32 + tm * 16 + r + hi * 8;
                float v = acc[tm][tn][r] + bv;
                if (out_f32) out_f32[(size_t)m * N + n] = v;
                else         out_bf[(size_t)m * N + n]  = f2bf(v);
            }
        }
}

// ---------------------------------------------------------------------------
// Flash-style attention. One workgroup (128 threads = 4 waves) handles one
// (b, h, 64-row q block). K/V 64-key blocks double-buffered in LDS via TDM,
// online softmax in registers, WMMA for both Q*K^T and P*V.
// qkv: bf16 [B, S, 3H]; y: bf16 [B, S, H]
// ---------------------------------------------------------------------------
__global__ __launch_bounds__(128)
void attn_kernel(const unsigned short* __restrict__ qkv,
                 const unsigned char* __restrict__ mask,
                 unsigned short* __restrict__ ybf) {
    constexpr int S  = 512;
    constexpr int H  = 768;
    constexpr int H3 = 3 * H;
    constexpr int NH = 12;

    __shared__ unsigned short Qs[64 * 64];
    __shared__ unsigned short Ks[2][64 * 64];
    __shared__ unsigned short Vs[2][64 * 64];
    __shared__ unsigned short Ps[64 * 64];

    int id = blockIdx.x;
    int qb = id & 7;
    int t2 = id >> 3;
    int h  = t2 % NH;
    int b  = t2 / NH;
    int q0 = qb * 64;

    const int tid  = threadIdx.x;
    const int lane = tid & 31;
    const int wave = tid >> 5;     // 0..3 -> q rows [wave*16, wave*16+16)
    const int hl   = lane & 15;
    const int hi   = (lane >> 4) & 1;

    const unsigned short* Qg = qkv + ((size_t)b * S + q0) * H3 + h * 64;
    const unsigned short* Kg = qkv + (size_t)b * S * H3 + H + h * 64;
    const unsigned short* Vg = qkv + (size_t)b * S * H3 + 2 * H + h * 64;
    const unsigned char*  mb = mask + (size_t)b * S;

#if USE_TDM
    if (wave == 0) {
        tdm_load_2d_bf16(Qg, Qs, 64, 64, H3);
        tdm_load_2d_bf16(Kg, &Ks[0][0], 64, 64, H3);
        tdm_load_2d_bf16(Vg, &Vs[0][0], 64, 64, H3);
    }
#else
    for (int i = tid; i < 64 * 8; i += 128) {
        int r = i >> 3, c = (i & 7) * 8;
        *(uint4*)(Qs + r * 64 + c) = *(const uint4*)(Qg + (size_t)r * H3 + c);
        *(uint4*)(&Ks[0][0] + r * 64 + c) =
            *(const uint4*)(Kg + (size_t)r * H3 + c);
        *(uint4*)(&Vs[0][0] + r * 64 + c) =
            *(const uint4*)(Vg + (size_t)r * H3 + c);
    }
#endif

    v8f o[4];
    float mrun[8], lrun[8];
#pragma unroll
    for (int t = 0; t < 4; ++t)
#pragma unroll
        for (int e = 0; e < 8; ++e) o[t][e] = 0.0f;
#pragma unroll
    for (int r = 0; r < 8; ++r) { mrun[r] = -3.0e38f; lrun[r] = 0.0f; }

    for (int kb = 0; kb < 8; ++kb) {
        const int buf = kb & 1;
#if USE_TDM
        if (wave == 0) __builtin_amdgcn_s_wait_tensorcnt(0);
        __syncthreads();
        if (kb + 1 < 8 && wave == 0) {
            tdm_load_2d_bf16(Kg + (size_t)(kb + 1) * 64 * H3, &Ks[buf ^ 1][0],
                             64, 64, H3);
            tdm_load_2d_bf16(Vg + (size_t)(kb + 1) * 64 * H3, &Vs[buf ^ 1][0],
                             64, 64, H3);
        }
#else
        __syncthreads();
        if (kb + 1 < 8) {
            for (int i = tid; i < 64 * 8; i += 128) {
                int r = i >> 3, c = (i & 7) * 8;
                *(uint4*)(&Ks[buf ^ 1][0] + r * 64 + c) =
                    *(const uint4*)(Kg + (size_t)((kb + 1) * 64 + r) * H3 + c);
                *(uint4*)(&Vs[buf ^ 1][0] + r * 64 + c) =
                    *(const uint4*)(Vg + (size_t)((kb + 1) * 64 + r) * H3 + c);
            }
        }
#endif

        // ---- scores: 16 q rows x 64 keys via WMMA (K dim = 64 -> 2 steps)
        v8f sacc[4];
#pragma unroll
        for (int t = 0; t < 4; ++t)
#pragma unroll
            for (int e = 0; e < 8; ++e) sacc[t][e] = 0.0f;

#pragma unroll
        for (int ks = 0; ks < 2; ++ks) {
            BF16x16 a;
#pragma unroll
            for (int j = 0; j < 8; ++j)
                a.u[j] = *(const unsigned int*)
                    (Qs + (wave * 16 + hl) * 64 + ks * 32 + a_koff(j, hi));
#pragma unroll
            for (int t = 0; t < 4; ++t) {
                BF16x16 bf;  // B = K^T : element (k=dk, n=key) = Ks[key*64+dk]
#pragma unroll
                for (int j = 0; j < 8; ++j)
                    bf.u[j] = *(const unsigned int*)
                        (&Ks[buf][0] + (t * 16 + hl) * 64 + ks * 32 + b_koff(j, hi));
                sacc[t] = __builtin_amdgcn_wmma_f32_16x16x32_bf16(
                    false, a.v, false, bf.v, (short)0, sacc[t], false, false);
            }
        }

        // ---- scale + mask (lane's column = one key)
#pragma unroll
        for (int t = 0; t < 4; ++t) {
            int kglob = kb * 64 + t * 16 + hl;
            bool ok = mb[kglob] != 0;
#pragma unroll
            for (int e = 0; e < 8; ++e)
                sacc[t][e] = ok ? sacc[t][e] * 0.125f : -3.0e38f;
        }

        // ---- online softmax: row max over 64 keys (4 tiles + 16-lane half)
        float mnew[8], alpha[8];
#pragma unroll
        for (int r = 0; r < 8; ++r) {
            float v = fmaxf(fmaxf(sacc[0][r], sacc[1][r]),
                            fmaxf(sacc[2][r], sacc[3][r]));
            v = fmaxf(v, __shfl_xor(v, 1));
            v = fmaxf(v, __shfl_xor(v, 2));
            v = fmaxf(v, __shfl_xor(v, 4));
            v = fmaxf(v, __shfl_xor(v, 8));
            float m2 = fmaxf(mrun[r], v);
            alpha[r] = __builtin_exp2f((mrun[r] - m2) * 1.44269504f);
            mnew[r]  = m2;
        }
#pragma unroll
        for (int t = 0; t < 4; ++t)
#pragma unroll
            for (int e = 0; e < 8; ++e)
                sacc[t][e] = __builtin_exp2f((sacc[t][e] - mnew[e]) * 1.44269504f);
#pragma unroll
        for (int r = 0; r < 8; ++r) {
            float s = sacc[0][r] + sacc[1][r] + sacc[2][r] + sacc[3][r];
            s += __shfl_xor(s, 1);
            s += __shfl_xor(s, 2);
            s += __shfl_xor(s, 4);
            s += __shfl_xor(s, 8);
            lrun[r] = lrun[r] * alpha[r] + s;
            mrun[r] = mnew[r];
        }
        // rescale running O
#pragma unroll
        for (int t = 0; t < 4; ++t)
#pragma unroll
            for (int e = 0; e < 8; ++e) o[t][e] *= alpha[e];

        // ---- write P (bf16) to LDS row-major, regather in A-layout
#pragma unroll
        for (int t = 0; t < 4; ++t)
#pragma unroll
            for (int e = 0; e < 8; ++e)
                Ps[(wave * 16 + e + hi * 8) * 64 + t * 16 + hl] = f2bf(sacc[t][e]);
        __syncthreads();

        // ---- O += P * V  (A = P 16x64, B = V 64x64)
#pragma unroll
        for (int ks = 0; ks < 2; ++ks) {
            BF16x16 a;
#pragma unroll
            for (int j = 0; j < 8; ++j)
                a.u[j] = *(const unsigned int*)
                    (Ps + (wave * 16 + hl) * 64 + ks * 32 + a_koff(j, hi));
#pragma unroll
            for (int t = 0; t < 4; ++t) {
                BF16x16 bf;  // B element (k=key, n=dk) = Vs[key*64 + dk]
#pragma unroll
                for (int j = 0; j < 8; ++j) {
                    int kk = ks * 32 + b_koff(j, hi);
                    unsigned int lo  = Vs[buf][(kk + 0) * 64 + t * 16 + hl];
                    unsigned int hi2 = Vs[buf][(kk + 1) * 64 + t * 16 + hl];
                    bf.u[j] = lo | (hi2 << 16);
                }
                o[t] = __builtin_amdgcn_wmma_f32_16x16x32_bf16(
                    false, a.v, false, bf.v, (short)0, o[t], false, false);
            }
        }
    }

    // ---- normalize and write Y (bf16, [B,S,H] layout)
    unsigned short* Yg = ybf + ((size_t)b * S + q0) * H + h * 64;
#pragma unroll
    for (int r = 0; r < 8; ++r) {
        float inv = 1.0f / lrun[r];
        int m = wave * 16 + r + hi * 8;
#pragma unroll
        for (int t = 0; t < 4; ++t)
            Yg[(size_t)m * H + t * 16 + hl] = f2bf(o[t][r] * inv);
    }
}

// ---------------------------------------------------------------------------
// Host launcher
// ---------------------------------------------------------------------------
extern "C" void kernel_launch(void* const* d_in, const int* in_sizes, int n_in,
                              void* d_out, int out_size, void* d_ws, size_t ws_size,
                              hipStream_t stream) {
    (void)in_sizes; (void)n_in; (void)out_size; (void)ws_size;

    constexpr int B = 16, S = 512, H = 768;
    constexpr int M = B * S;        // 8192
    constexpr int N_QKV = 3 * H;    // 2304

    const float*         x     = (const float*)d_in[0];
    const unsigned char* mask  = (const unsigned char*)d_in[1];
    const float*         wqkv  = (const float*)d_in[2];
    const float*         bqkv  = (const float*)d_in[3];
    const float*         wo    = (const float*)d_in[4];
    const float*         bo    = (const float*)d_in[5];
    float*               out   = (float*)d_out;

    auto align256 = [](size_t v) { return (v + 255) & ~(size_t)255; };
    char* ws = (char*)d_ws;
    unsigned short* Xbf   = (unsigned short*)ws; ws += align256((size_t)M * H * 2);
    unsigned short* Wqkvb = (unsigned short*)ws; ws += align256((size_t)N_QKV * H * 2);
    unsigned short* Wob   = (unsigned short*)ws; ws += align256((size_t)H * H * 2);
    unsigned short* QKVb  = (unsigned short*)ws; ws += align256((size_t)M * N_QKV * 2);
    unsigned short* Ybf   = (unsigned short*)ws; ws += align256((size_t)M * H * 2);

    // 1) convert fp32 inputs to bf16
    int nx = M * H;
    f32_to_bf16_kernel<<<(nx + 255) / 256, 256, 0, stream>>>(x, Xbf, nx);
    int nwq = N_QKV * H;
    f32_to_bf16_kernel<<<(nwq + 255) / 256, 256, 0, stream>>>(wqkv, Wqkvb, nwq);
    int nwo = H * H;
    f32_to_bf16_kernel<<<(nwo + 255) / 256, 256, 0, stream>>>(wo, Wob, nwo);

    // 2) QKV projection: [8192,768] x [2304,768]^T + b -> bf16 [B,S,3H]
    gemm_bf16_kernel<<<(M / 128) * (N_QKV / 64), 256, 0, stream>>>(
        Xbf, Wqkvb, bqkv, QKVb, nullptr, M, N_QKV, H);

    // 3) attention: per (b, head, 64-row q block)
    attn_kernel<<<B * 12 * (S / 64), 128, 0, stream>>>(QKVb, mask, Ybf);

    // 4) output projection: [8192,768] x [768,768]^T + b -> fp32 d_out
    gemm_bf16_kernel<<<(M / 128) * (H / 64), 256, 0, stream>>>(
        Ybf, Wob, bo, nullptr, out, M, H, H);
}
